// DiamondAttention_23596550324272
// MI455X (gfx1250) — matrix operand order repaired and checked by reading, not compiled, not run on hardware
//
#include <hip/hip_runtime.h>
#include <hip/hip_bf16.h>
#include <stdint.h>

// ---------------- CDNA5 WMMA plumbing ----------------
typedef __attribute__((ext_vector_type(16))) __bf16 v16bf;
typedef __attribute__((ext_vector_type(8)))  float  v8f;

union FragAB {
  v16bf v;
  uint4 u[2];
  __bf16 h[16];
};
union FragC {
  v8f   v;
  float f[8];
};

#define WMMA_BF16(a, b, c) \
  __builtin_amdgcn_wmma_f32_16x16x32_bf16(false, (a), false, (b), (short)0, (c), false, false)

static __device__ __forceinline__ float silu_f(float x) {
  return x / (1.0f + __expf(-x));
}

// Problem constants
#define BB 4
#define NN 2048
#define HH 512
#define NHH 8
#define HDD 64
#define MM (BB * NN)   // 8192

// =====================================================================
// One-time converts (paid once per launch, not per GEMM block/K-step)
// =====================================================================

// elementwise f32 -> bf16, 8 elements / thread
__global__ __launch_bounds__(256) void cvt_kernel(
    const float* __restrict__ in, __bf16* __restrict__ out, int n)
{
  const int i = (blockIdx.x * 256 + threadIdx.x) * 8;
  if (i >= n) return;
  const float4 a = *(const float4*)(in + i);
  const float4 b = *(const float4*)(in + i + 4);
  union { __bf16 h[8]; uint4 u; } r;
  r.h[0] = (__bf16)a.x; r.h[1] = (__bf16)a.y;
  r.h[2] = (__bf16)a.z; r.h[3] = (__bf16)a.w;
  r.h[4] = (__bf16)b.x; r.h[5] = (__bf16)b.y;
  r.h[6] = (__bf16)b.z; r.h[7] = (__bf16)b.w;
  *(uint4*)(out + i) = r.u;
}

// W[k][n] f32 -> WT[n][k] bf16 (32x32 LDS tile transpose; optional batch z)
__global__ __launch_bounds__(256) void transcvt_kernel(
    const float* __restrict__ W, __bf16* __restrict__ WT, int K, int N,
    size_t inBStride, size_t outBStride)
{
  W  += (size_t)blockIdx.z * inBStride;
  WT += (size_t)blockIdx.z * outBStride;
  __shared__ float t[32][33];
  const int k0 = blockIdx.x * 32, n0 = blockIdx.y * 32;
  const int tx = threadIdx.x & 31, ty = threadIdx.x >> 5;  // 32 x 8
  #pragma unroll
  for (int r = 0; r < 32; r += 8)
    t[ty + r][tx] = W[(size_t)(k0 + ty + r) * N + n0 + tx];
  __syncthreads();
  #pragma unroll
  for (int r = 0; r < 32; r += 8)
    WT[(size_t)(n0 + ty + r) * K + k0 + tx] = (__bf16)t[tx][ty + r];
}

// =====================================================================
// LDS-free bf16 WMMA GEMM:  C[M,N] = act( [A0 | A1] @ B + bias )
//   A0/A1: bf16 row-major (widths K0 / Ktot-K0); WT: bf16 [Ndim x Ktot]
//   Fragments loaded directly from global (hot in L2/WGP$).
//   OUTMODE: 0 = bf16 row-major, 1 = f32 row-major,
//            2 = bf16 transposed per batch: out[(b*HH + col)*NN + (row&2047)]
//   BATCHB : B is per-batch [BB][Ndim][Ktot], batch = row >> 11
// =====================================================================
template<bool SILU, int OUTMODE, bool BATCHB>
__global__ __launch_bounds__(256) void gemm_direct(
    const __bf16* __restrict__ A0, const __bf16* __restrict__ A1,
    int K0, int Ktot,
    const __bf16* __restrict__ WT, const float* __restrict__ bias,
    float* __restrict__ outF, __bf16* __restrict__ outB, int Ndim)
{
  const int tid  = threadIdx.x;
  const int m0   = blockIdx.y * 128;
  const int n0   = blockIdx.x * 128;
  const int wave = tid >> 5, lane = tid & 31;
  const int hl   = lane >> 4, l16 = lane & 15;
  const int wm   = wave & 3,  wn  = wave >> 2;
  const int K1   = Ktot - K0;
  const size_t bboff = BATCHB ? (size_t)(m0 >> 11) * Ndim * Ktot : 0;

  FragC acc[2][4];
  #pragma unroll
  for (int mi = 0; mi < 2; mi++)
    #pragma unroll
    for (int ni = 0; ni < 4; ni++)
      #pragma unroll
      for (int i = 0; i < 8; i++) acc[mi][ni].f[i] = 0.f;

  for (int k0 = 0; k0 < Ktot; k0 += 32) {
    const bool lo = (k0 < K0);
    const __bf16* Ab = lo ? A0 : A1;
    const int kw = lo ? K0 : K1;
    const int ko = lo ? k0 : (k0 - K0);

    FragAB a[2];
    #pragma unroll
    for (int mi = 0; mi < 2; mi++) {
      const int row = m0 + wm * 32 + mi * 16 + l16;
      const __bf16* ap = Ab + (size_t)row * kw + ko + hl * 8;
      a[mi].u[0] = *(const uint4*)ap;
      a[mi].u[1] = *(const uint4*)(ap + 16);
    }
    #pragma unroll
    for (int ni = 0; ni < 4; ni++) {
      FragAB bf;
      const __bf16* bp = WT + bboff
          + (size_t)(n0 + wn * 64 + ni * 16 + l16) * Ktot + k0 + hl * 8;
      bf.u[0] = *(const uint4*)bp;
      bf.u[1] = *(const uint4*)(bp + 16);
      #pragma unroll
      for (int mi = 0; mi < 2; mi++)
        acc[mi][ni].v = WMMA_BF16(a[mi].v, bf.v, acc[mi][ni].v);
    }
  }

  #pragma unroll
  for (int mi = 0; mi < 2; mi++)
    #pragma unroll
    for (int ni = 0; ni < 4; ni++) {
      const int col = n0 + wn * 64 + ni * 16 + l16;
      const float bsv = bias ? bias[col] : 0.f;
      #pragma unroll
      for (int i = 0; i < 8; i++) {
        const int row = m0 + wm * 32 + mi * 16 + hl * 8 + i;
        float x = acc[mi][ni].f[i] + bsv;
        if (SILU) x = silu_f(x);
        if (OUTMODE == 1) {
          outF[(size_t)row * Ndim + col] = x;
        } else if (OUTMODE == 2) {
          const int bb = row >> 11, key = row & 2047;
          outB[((size_t)(bb * HH) + col) * NN + key] = (__bf16)x;
        } else {
          outB[(size_t)row * Ndim + col] = (__bf16)x;
        }
      }
    }
}

// =====================================================================
// Attention: per (b, head, 128-query tile). Two-pass flash softmax.
//   Q/K frags stream directly from global; V streams from the transposed
//   VbT[B,H,N] layout; only the P (C->A) re-fragmentation uses LDS.
// =====================================================================
__global__ __launch_bounds__(256) void attn_kernel(
    const __bf16* __restrict__ Qb, const __bf16* __restrict__ Kb,
    const __bf16* __restrict__ VbT, float* __restrict__ Wt,
    __bf16* __restrict__ Cxb)
{
  const float scale = 0.125f;  // 1/sqrt(64)
  const float hinv  = 0.125f;  // 1/NH

  __shared__ __align__(16) __bf16 Ptile[8 * 16 * 128];  // per-wave P tiles

  const int qt = blockIdx.x, hh = blockIdx.y, b = blockIdx.z;
  const int tid = threadIdx.x, wave = tid >> 5, lane = tid & 31;
  const int hl = lane >> 4, l16 = lane & 15;

  // Q A-fragments (persist across both passes)
  FragAB qa[2];
  {
    const int qrow = qt * 128 + wave * 16 + l16;
    const __bf16* qp = Qb + ((size_t)(b * NN + qrow)) * HH + hh * HDD;
    #pragma unroll
    for (int ks = 0; ks < 2; ks++) {
      const __bf16* p = qp + ks * 32 + hl * 8;
      qa[ks].u[0] = *(const uint4*)p;
      qa[ks].u[1] = *(const uint4*)(p + 16);
    }
  }

  float rm[8], rl[8];
  #pragma unroll
  for (int i = 0; i < 8; i++) { rm[i] = -3.0e38f; rl[i] = 0.f; }

  // ---------------- pass 1: online row max / sum-exp ----------------
  for (int kt = 0; kt < 16; kt++) {
    const int kb = kt * 128;
    FragC s[8];
    #pragma unroll
    for (int nt = 0; nt < 8; nt++) {
      #pragma unroll
      for (int i = 0; i < 8; i++) s[nt].f[i] = 0.f;
      #pragma unroll
      for (int ks = 0; ks < 2; ks++) {
        FragAB kf;
        const __bf16* kp = Kb + ((size_t)(b * NN + kb + nt * 16 + l16)) * HH
                              + hh * HDD + ks * 32 + hl * 8;
        kf.u[0] = *(const uint4*)kp;
        kf.u[1] = *(const uint4*)(kp + 16);
        s[nt].v = WMMA_BF16(qa[ks].v, kf.v, s[nt].v);
      }
    }
    #pragma unroll
    for (int i = 0; i < 8; i++) {
      float v = -3.0e38f;
      #pragma unroll
      for (int nt = 0; nt < 8; nt++) v = fmaxf(v, s[nt].f[i]);
      float tm = v * scale;
      #pragma unroll
      for (int off = 8; off >= 1; off >>= 1)
        tm = fmaxf(tm, __shfl_xor(tm, off, 32));
      const float mn = fmaxf(rm[i], tm);
      float su = 0.f;
      #pragma unroll
      for (int nt = 0; nt < 8; nt++)
        su += __expf(s[nt].f[i] * scale - mn);
      #pragma unroll
      for (int off = 8; off >= 1; off >>= 1)
        su += __shfl_xor(su, off, 32);
      rl[i] = rl[i] * __expf(rm[i] - mn) + su;
      rm[i] = mn;
    }
  }

  // ---------------- pass 2: probabilities, weights, ctx ----------------
  FragC cacc[4];
  #pragma unroll
  for (int dt = 0; dt < 4; dt++)
    #pragma unroll
    for (int i = 0; i < 8; i++) cacc[dt].f[i] = 0.f;

  for (int kt = 0; kt < 16; kt++) {
    const int kb = kt * 128;
    // recompute scores -> exact probabilities
    #pragma unroll
    for (int nt = 0; nt < 8; nt++) {
      FragC s;
      #pragma unroll
      for (int i = 0; i < 8; i++) s.f[i] = 0.f;
      #pragma unroll
      for (int ks = 0; ks < 2; ks++) {
        FragAB kf;
        const __bf16* kp = Kb + ((size_t)(b * NN + kb + nt * 16 + l16)) * HH
                              + hh * HDD + ks * 32 + hl * 8;
        kf.u[0] = *(const uint4*)kp;
        kf.u[1] = *(const uint4*)(kp + 16);
        s.v = WMMA_BF16(qa[ks].v, kf.v, s.v);
      }
      #pragma unroll
      for (int i = 0; i < 8; i++) {
        const float p = __expf(s.f[i] * scale - rm[i]) / rl[i];
        const int q    = qt * 128 + wave * 16 + hl * 8 + i;
        const int kcol = kb + nt * 16 + l16;
        atomicAdd(&Wt[((size_t)b * NN + q) * NN + kcol], p * hinv);
        Ptile[wave * 2048 + (hl * 8 + i) * 128 + nt * 16 + l16] = (__bf16)p;
      }
    }
    // ctx += P @ V : P refragmented via per-wave LDS (same-wave LDS ops are
    // in-order per ISA, so no barrier needed); V streamed from VbT[B,H,N].
    #pragma unroll
    for (int ks = 0; ks < 4; ks++) {
      FragAB pa;
      const __bf16* pp = &Ptile[wave * 2048 + l16 * 128 + ks * 32 + hl * 8];
      pa.u[0] = *(const uint4*)pp;
      pa.u[1] = *(const uint4*)(pp + 16);
      #pragma unroll
      for (int dt = 0; dt < 4; dt++) {
        FragAB vf;
        const __bf16* vp =
            VbT + ((size_t)(b * HH + hh * HDD + dt * 16 + l16)) * NN
                + kb + ks * 32 + hl * 8;
        vf.u[0] = *(const uint4*)vp;
        vf.u[1] = *(const uint4*)(vp + 16);
        cacc[dt].v = WMMA_BF16(pa.v, vf.v, cacc[dt].v);
      }
    }
  }

  // store ctx (bf16, row-major [B,N,H])
  #pragma unroll
  for (int dt = 0; dt < 4; dt++)
    #pragma unroll
    for (int i = 0; i < 8; i++) {
      const int q = qt * 128 + wave * 16 + hl * 8 + i;
      Cxb[((size_t)(b * NN + q)) * HH + hh * HDD + dt * 16 + l16] =
          (__bf16)cacc[dt].f[i];
    }
}

// =====================================================================
// Host orchestration
// =====================================================================
extern "C" void kernel_launch(void* const* d_in, const int* in_sizes, int n_in,
                              void* d_out, int out_size, void* d_ws, size_t ws_size,
                              hipStream_t stream) {
  (void)in_sizes; (void)n_in; (void)out_size; (void)ws_size;

  const float* query  = (const float*)d_in[0];
  const float* key    = (const float*)d_in[1];
  const float* lvalue = (const float*)d_in[2];
  const float* rvalue = (const float*)d_in[3];
  const float *Wq = (const float*)d_in[4],  *bq = (const float*)d_in[5];
  const float *Wk = (const float*)d_in[6],  *bk = (const float*)d_in[7];
  const float *Wv = (const float*)d_in[8],  *bv = (const float*)d_in[9];
  const float *Wo = (const float*)d_in[10], *bo = (const float*)d_in[11];
  const float *Wl = (const float*)d_in[12], *bl = (const float*)d_in[13];
  const float *Wl1 = (const float*)d_in[14], *bl1 = (const float*)d_in[15];
  const float *Wl2 = (const float*)d_in[16], *bl2 = (const float*)d_in[17];
  const float *Wr1 = (const float*)d_in[18], *br1 = (const float*)d_in[19];
  const float *Wr2 = (const float*)d_in[20], *br2 = (const float*)d_in[21];

  float* out_left  = (float*)d_out;                  // [B,N,H]
  float* out_right = out_left + (size_t)MM * HH;     // [B,N,H]
  float* out_w     = out_right + (size_t)MM * HH;    // [B,N,N]

  char* ws = (char*)d_ws;
  const size_t SZ = (size_t)MM * HH * sizeof(__bf16);  // 8 MB
  size_t off = 0;
  __bf16* Qin  = (__bf16*)(ws + off); off += SZ;
  __bf16* Kin  = (__bf16*)(ws + off); off += SZ;
  __bf16* Lin  = (__bf16*)(ws + off); off += SZ;
  __bf16* Rin  = (__bf16*)(ws + off); off += SZ;
  __bf16* Qb   = (__bf16*)(ws + off); off += SZ;
  __bf16* Kb   = (__bf16*)(ws + off); off += SZ;
  __bf16* VbT  = (__bf16*)(ws + off); off += SZ;      // [B,H,N]
  __bf16* LPbT = (__bf16*)(ws + off); off += SZ;      // [B,H,N]
  __bf16* Cxb  = (__bf16*)(ws + off); off += SZ;
  __bf16* ARb  = (__bf16*)(ws + off); off += SZ;
  __bf16* ALb  = (__bf16*)(ws + off); off += SZ;
  __bf16* Ub   = (__bf16*)(ws + off); off += SZ;
  __bf16* WtT  = (__bf16*)(ws + off);                 // [B,N,N] bf16, 32 MB
  off += (size_t)BB * NN * NN * sizeof(__bf16);
  const size_t WSZ  = (size_t)HH * HH * sizeof(__bf16);      // 512 KB
  const size_t WSZ2 = (size_t)HH * 2 * HH * sizeof(__bf16);  // 1 MB
  __bf16* WTq  = (__bf16*)(ws + off); off += WSZ;
  __bf16* WTk  = (__bf16*)(ws + off); off += WSZ;
  __bf16* WTv  = (__bf16*)(ws + off); off += WSZ;
  __bf16* WTo  = (__bf16*)(ws + off); off += WSZ;
  __bf16* WTl  = (__bf16*)(ws + off); off += WSZ;
  __bf16* WTl2 = (__bf16*)(ws + off); off += WSZ;
  __bf16* WTr2 = (__bf16*)(ws + off); off += WSZ;
  __bf16* WTl1 = (__bf16*)(ws + off); off += WSZ2;
  __bf16* WTr1 = (__bf16*)(ws + off); off += WSZ2;

  // weights output accumulated via atomics -> zero every call
  hipMemsetAsync(out_w, 0, (size_t)BB * NN * NN * sizeof(float), stream);

  const dim3 blk(256);
  const int nAct = MM * HH;
  const dim3 gcvt((nAct / 8 + 255) / 256);
  cvt_kernel<<<gcvt, blk, 0, stream>>>(query,  Qin, nAct);
  cvt_kernel<<<gcvt, blk, 0, stream>>>(key,    Kin, nAct);
  cvt_kernel<<<gcvt, blk, 0, stream>>>(lvalue, Lin, nAct);
  cvt_kernel<<<gcvt, blk, 0, stream>>>(rvalue, Rin, nAct);

  const dim3 gt1(HH / 32, HH / 32);          // 512x512 weights
  const dim3 gt2(2 * HH / 32, HH / 32);      // 1024x512 weights
  transcvt_kernel<<<gt1, blk, 0, stream>>>(Wq,  WTq,  HH, HH, 0, 0);
  transcvt_kernel<<<gt1, blk, 0, stream>>>(Wk,  WTk,  HH, HH, 0, 0);
  transcvt_kernel<<<gt1, blk, 0, stream>>>(Wv,  WTv,  HH, HH, 0, 0);
  transcvt_kernel<<<gt1, blk, 0, stream>>>(Wo,  WTo,  HH, HH, 0, 0);
  transcvt_kernel<<<gt1, blk, 0, stream>>>(Wl,  WTl,  HH, HH, 0, 0);
  transcvt_kernel<<<gt1, blk, 0, stream>>>(Wl2, WTl2, HH, HH, 0, 0);
  transcvt_kernel<<<gt1, blk, 0, stream>>>(Wr2, WTr2, HH, HH, 0, 0);
  transcvt_kernel<<<gt2, blk, 0, stream>>>(Wl1, WTl1, 2 * HH, HH, 0, 0);
  transcvt_kernel<<<gt2, blk, 0, stream>>>(Wr1, WTr1, 2 * HH, HH, 0, 0);

  const dim3 g512(HH / 128, MM / 128);        // (4, 64)

  // input projections
  gemm_direct<false, 0, false><<<g512, blk, 0, stream>>>(
      Qin, Qin, HH, HH, WTq, bq, nullptr, Qb, HH);
  gemm_direct<false, 0, false><<<g512, blk, 0, stream>>>(
      Kin, Kin, HH, HH, WTk, bk, nullptr, Kb, HH);
  gemm_direct<false, 2, false><<<g512, blk, 0, stream>>>(
      Rin, Rin, HH, HH, WTv, bv, nullptr, VbT, HH);     // transposed V out
  gemm_direct<false, 2, false><<<g512, blk, 0, stream>>>(
      Lin, Lin, HH, HH, WTl, bl, nullptr, LPbT, HH);    // transposed LP out

  // attention: ctx + head-averaged weights
  const dim3 ga(NN / 128, NHH, BB);           // (16, 8, 4)
  attn_kernel<<<ga, blk, 0, stream>>>(Qb, Kb, VbT, out_w, Cxb);

  // attn_right = ctx @ Wo + bo
  gemm_direct<false, 0, false><<<g512, blk, 0, stream>>>(
      Cxb, Cxb, HH, HH, WTo, bo, nullptr, ARb, HH);

  // attn_left = weights^T @ lproj:
  //   1) transpose+convert weights: [B,q,m] f32 -> WtT [B,m,q] bf16
  //   2) streaming GEMM with per-batch B = LPbT[B,H,N]
  const dim3 gtw(NN / 32, NN / 32, BB);
  transcvt_kernel<<<gtw, blk, 0, stream>>>(out_w, WtT, NN, NN,
                                           (size_t)NN * NN, (size_t)NN * NN);
  gemm_direct<false, 0, true><<<g512, blk, 0, stream>>>(
      WtT, WtT, NN, NN, LPbT, nullptr, nullptr, ALb, HH);

  // left  = silu([attn_left | rvalue] @ Wl1 + bl1) @ Wl2 + bl2
  gemm_direct<true, 0, false><<<g512, blk, 0, stream>>>(
      ALb, Rin, HH, 2 * HH, WTl1, bl1, nullptr, Ub, HH);
  gemm_direct<false, 1, false><<<g512, blk, 0, stream>>>(
      Ub, Ub, HH, HH, WTl2, bl2, out_left, nullptr, HH);

  // right = silu([attn_right | lvalue] @ Wr1 + br1) @ Wr2 + br2
  gemm_direct<true, 0, false><<<g512, blk, 0, stream>>>(
      ARb, Lin, HH, 2 * HH, WTr1, br1, nullptr, Ub, HH);
  gemm_direct<false, 1, false><<<g512, blk, 0, stream>>>(
      Ub, Ub, HH, HH, WTr2, br2, out_right, nullptr, HH);
}
